// NAM_48533130444845
// MI455X (gfx1250) — compile-verified, matrix operand order checked by reading, not run on hardware
//
#include <hip/hip_runtime.h>
#include <hip/hip_bf16.h>

typedef __attribute__((ext_vector_type(16))) _Float16 v16h;
typedef __attribute__((ext_vector_type(8)))  float    v8f;

#define B_TOT 8192
#define F_TOT 128
#define H_DIM 64
#define ROWS_PER_WAVE 16          // M per WMMA tile
#define MT_PER_WAVE 4             // M subtiles per wave (B-fragment reuse)
#define WAVES_PER_BLOCK 8
#define ROWS_PER_BLOCK (ROWS_PER_WAVE * MT_PER_WAVE * WAVES_PER_BLOCK)  // 512
#define W2T_STRIDE 72   // halves; 144B row pitch -> conflict-free 16-lane reads

// Fused NAM kernel: layer1 (scalar->H, fused into A-frag build),
// layer2 (H x H batched GEMM via v_wmma_f32_16x16x32_f16, B frags register-
// resident and reused across 4 M-subtiles to stay under LDS bandwidth),
// layer3 (H->scalar, fused epilogue w/ cross-lane butterfly reduce).
__global__ __launch_bounds__(256) void nam_fused_kernel(
    const float* __restrict__ x,    // (B,F)
    const float* __restrict__ W1,   // (F,H)
    const float* __restrict__ b1,   // (F,H)
    const float* __restrict__ W2,   // (F,H,H) [h = contraction, k = out]
    const float* __restrict__ b2,   // (F,H)
    const float* __restrict__ W3,   // (F,H)
    const float* __restrict__ b3,   // (F,)
    float* __restrict__ fnn)        // (B,F)
{
    __shared__ _Float16 sW2T[H_DIM * W2T_STRIDE];  // [n][h] transposed, f16
    __shared__ float sW1[H_DIM], sB1[H_DIM], sB2[H_DIM], sW3[H_DIM];

    const int f       = blockIdx.y;
    const int rowBase = blockIdx.x * ROWS_PER_BLOCK;
    const int t       = threadIdx.x;

    // ---- Stage weights for this feature into LDS (once per block) ----
    const float* __restrict__ W2f = W2 + (size_t)f * (H_DIM * H_DIM);
#pragma unroll
    for (int i = 0; i < (H_DIM * H_DIM) / 256; ++i) {
        int idx = i * 256 + t;             // coalesced global read
        int h = idx >> 6, n = idx & 63;
        sW2T[n * W2T_STRIDE + h] = (_Float16)W2f[idx];
    }
    if (t < H_DIM) {
        sW1[t] = W1[f * H_DIM + t];
        sB1[t] = b1[f * H_DIM + t];
        sB2[t] = b2[f * H_DIM + t];
        sW3[t] = W3[f * H_DIM + t];
    }
    __syncthreads();

    const int wave = t >> 5;
    const int lane = t & 31;
    const int mrow = lane & 15;            // M (and fragment N) sub-index
    const int koff = (lane >> 4) << 3;     // 0 or 8: K half-select per ISA layout
    const int waveRow = rowBase + wave * (ROWS_PER_WAVE * MT_PER_WAVE);

    const float b3v = b3[f];

    // ---- Prefetch the 4 per-subtile inputs (overlap with B-frag loads) ----
    float xv[MT_PER_WAVE];
#pragma unroll
    for (int mt = 0; mt < MT_PER_WAVE; ++mt)
        xv[mt] = x[(size_t)(waveRow + mt * ROWS_PER_WAVE + mrow) * F_TOT + f];

    // ---- Load all 8 B fragments once; register-resident across M tiles ----
    // B layout (transpose-symmetric to A): lane -> N = nt*16 + mrow,
    // K pairs {0..7,16..23} (lane<16) / {8..15,24..31} (lane>=16) + 32*kt.
    v16h bfr[2][4];
#pragma unroll
    for (int kt = 0; kt < 2; ++kt) {
#pragma unroll
        for (int nt = 0; nt < 4; ++nt) {
            const int n = nt * 16 + mrow;
            const _Float16* __restrict__ bp =
                &sW2T[n * W2T_STRIDE + kt * 32 + koff];
#pragma unroll
            for (int j = 0; j < 8; ++j) {
                int h0 = 2 * j + (j >= 4 ? 8 : 0);   // pairs -> ds_load_b128 merge
                bfr[kt][nt][2 * j]     = bp[h0];
                bfr[kt][nt][2 * j + 1] = bp[h0 + 1];
            }
        }
    }

    // ---- 4 M-subtiles x (M16 N64 K64) = 32 WMMAs per wave ----
#pragma unroll
    for (int mt = 0; mt < MT_PER_WAVE; ++mt) {
        // Layer 1 fused: build A fragments (16-bit A 16x32 layout)
        v16h a[2];
#pragma unroll
        for (int i = 0; i < 16; ++i) {
            int kb = koff + (i < 8 ? i : i + 8);
            float v0 = xv[mt] * sW1[kb]      + sB1[kb];
            float v1 = xv[mt] * sW1[kb + 32] + sB1[kb + 32];
            a[0][i] = (_Float16)(v0 > 0.f ? v0 : 0.f);
            a[1][i] = (_Float16)(v1 > 0.f ? v1 : 0.f);
        }

        v8f c[4];
#pragma unroll
        for (int nt = 0; nt < 4; ++nt) c[nt] = (v8f){};
#pragma unroll
        for (int kt = 0; kt < 2; ++kt)
#pragma unroll
            for (int nt = 0; nt < 4; ++nt)
                c[nt] = __builtin_amdgcn_wmma_f32_16x16x32_f16(
                    false, a[kt], false, bfr[kt][nt], (short)0, c[nt],
                    false, false);

        // Layer 3 fused epilogue.
        // C layout: lane -> N = nt*16 + mrow; VGPR v -> M = v + 8*(lane>>4)
        float acc[8];
#pragma unroll
        for (int v = 0; v < 8; ++v) acc[v] = 0.f;
#pragma unroll
        for (int nt = 0; nt < 4; ++nt) {
            const int n = nt * 16 + mrow;
            const float bb = sB2[n];
            const float ww = sW3[n];
#pragma unroll
            for (int v = 0; v < 8; ++v) {
                float h2 = c[nt][v] + bb;
                h2 = h2 > 0.f ? h2 : 0.f;
                acc[v] += h2 * ww;
            }
        }
        // reduce over the 16 lanes sharing the same M (stays within halves)
#pragma unroll
        for (int v = 0; v < 8; ++v) {
#pragma unroll
            for (int m = 1; m < 16; m <<= 1)
                acc[v] += __shfl_xor(acc[v], m, 32);
        }
        if (mrow == 0) {
            const int mbase = (lane >> 4) * 8;
#pragma unroll
            for (int v = 0; v < 8; ++v) {
                int r = waveRow + mt * ROWS_PER_WAVE + mbase + v;
                fnn[(size_t)r * F_TOT + f] = acc[v] + b3v;
            }
        }
    }
}

// out[b] = sum_f fnn[b,f]; one wave per row, coalesced 128B loads.
__global__ __launch_bounds__(256) void nam_reduce_kernel(
    const float* __restrict__ fnn, float* __restrict__ out)
{
    const int wave = threadIdx.x >> 5;
    const int lane = threadIdx.x & 31;
    const int row  = blockIdx.x * WAVES_PER_BLOCK + wave;
    const float* __restrict__ p = fnn + (size_t)row * F_TOT;
    float s = p[lane] + p[lane + 32] + p[lane + 64] + p[lane + 96];
#pragma unroll
    for (int m = 1; m < 32; m <<= 1) s += __shfl_xor(s, m, 32);
    if (lane == 0) out[row] = s;
}

extern "C" void kernel_launch(void* const* d_in, const int* in_sizes, int n_in,
                              void* d_out, int out_size, void* d_ws, size_t ws_size,
                              hipStream_t stream) {
    (void)in_sizes; (void)n_in; (void)out_size; (void)d_ws; (void)ws_size;

    const float* x  = (const float*)d_in[0];
    const float* W1 = (const float*)d_in[1];
    const float* b1 = (const float*)d_in[2];
    const float* W2 = (const float*)d_in[3];
    const float* b2 = (const float*)d_in[4];
    const float* W3 = (const float*)d_in[5];
    const float* b3 = (const float*)d_in[6];

    float* out = (float*)d_out;        // (B,)   first in tuple order
    float* fnn = out + B_TOT;          // (B,F)  second in tuple order

    dim3 grid1(B_TOT / ROWS_PER_BLOCK, F_TOT);   // 16 x 128
    nam_fused_kernel<<<grid1, 256, 0, stream>>>(x, W1, b1, W2, b2, W3, b3, fnn);

    nam_reduce_kernel<<<B_TOT / WAVES_PER_BLOCK, 256, 0, stream>>>(fnn, out);
}
